// M1_27968827032305
// MI455X (gfx1250) — compile-verified, matrix operand order checked by reading, not run on hardware
//
#include <hip/hip_runtime.h>

typedef __attribute__((ext_vector_type(16))) _Float16 v16h;
typedef __attribute__((ext_vector_type(8)))  _Float16 v8h;   // 16 bytes = b128
typedef __attribute__((ext_vector_type(8)))  float    v8f;

// ---------------------------------------------------------------- utilities
__global__ void fill_f32(float* p, float v, long n) {
  long i = (long)blockIdx.x * blockDim.x + threadIdx.x;
  if (i < n) p[i] = v;
}

// f32 -> f16 elementwise pack (activations)
__global__ void pack_f16(const float* __restrict__ in, _Float16* __restrict__ out, long n) {
  long i = (long)blockIdx.x * blockDim.x + threadIdx.x;
  if (i < n) out[i] = (_Float16)in[i];
}

// W[K,O] f32 -> Wt[O,K] f16 (transposed so B-fragments are contiguous)
__global__ void pack_wt(const float* __restrict__ W, _Float16* __restrict__ Wt, int K, int O) {
  long i = (long)blockIdx.x * blockDim.x + threadIdx.x;
  if (i >= (long)K * O) return;
  int k = (int)(i / O), o = (int)(i % O);
  Wt[(long)o * K + k] = (_Float16)W[i];
}

// deg[i] starts at 1.0 (self loop); add 1 per incoming edge (row = edge_index[1])
__global__ void deg_accum(const long long* __restrict__ ei, float* deg, int E) {
  int e = blockIdx.x * blockDim.x + threadIdx.x;
  if (e < E) atomicAdd(&deg[(int)ei[(long)E + e]], 1.0f);
}

__global__ void rsqrt_k(const float* __restrict__ deg, float* dinv, int N) {
  int i = blockIdx.x * blockDim.x + threadIdx.x;
  if (i < N) dinv[i] = rsqrtf(deg[i]);
}

__global__ void edge_norm(const long long* __restrict__ ei,
                          const float* __restrict__ dinv, float* nrm, int E) {
  int e = blockIdx.x * blockDim.x + threadIdx.x;
  if (e < E) nrm[e] = dinv[(int)ei[(long)E + e]] * dinv[(int)ei[e]];
}

// ---------------------------------------------------------------- WMMA GEMM
// Fragment loader: 16 halves = two contiguous b128 loads (ISA 16-bit A/B layout:
// halves j<8 at K = klo..klo+7, halves j>=8 at K = khi..khi+7 for this lane).
__device__ __forceinline__ v16h ld_frag(const _Float16* __restrict__ p, int klo, int khi) {
  v8h lo = *(const v8h*)(p + klo);
  v8h hi = *(const v8h*)(p + khi);
  return __builtin_shufflevector(lo, hi, 0, 1, 2, 3, 4, 5, 6, 7,
                                 8, 9, 10, 11, 12, 13, 14, 15);
}

// C[M,O] = A[M,K] * W[K,O]; A is f16 row-major [M,K], Wt is f16 transposed [O,K].
// One wave computes a 16(M) x 64(N) strip: 4 accumulators, A-fragment reused
// across 4 WMMAs per k-step. grid = (M/16, O/64), block = 32 (one wave).
__global__ void gemm_wmma_packed(const _Float16* __restrict__ A,
                                 const _Float16* __restrict__ Wt,
                                 float* __restrict__ C, int K, int O) {
  const int lane = threadIdx.x;        // 0..31
  const int rc   = lane & 15;          // A row / B-D column within tile
  const int hi8  = (lane >> 4) << 3;   // K half-select (+8 for lanes 16..31)
  const long m0  = (long)blockIdx.x * 16;
  const int  n0  = blockIdx.y * 64;
  const _Float16* arow = A + (m0 + rc) * (long)K;
  const _Float16* bcol = Wt + (long)(n0 + rc) * K;   // tile t adds t*16*K
  const long bstep = (long)16 * K;

  v8f acc[4] = {v8f{}, v8f{}, v8f{}, v8f{}};
  for (int kk = 0; kk < K; kk += 32) {
    const int klo = kk + hi8, khi = kk + 16 + hi8;
    __builtin_prefetch(arow + kk + 32, 0, 0);        // global_prefetch_b8
    v16h a = ld_frag(arow, klo, khi);
#pragma unroll
    for (int t = 0; t < 4; ++t) {
      v16h b = ld_frag(bcol + t * bstep, klo, khi);
      acc[t] = __builtin_amdgcn_wmma_f32_16x16x32_f16(
          /*neg_a=*/false, a, /*neg_b=*/false, b,
          /*c_mod=*/(short)0, acc[t], /*reuse_a=*/false, /*reuse_b=*/false);
    }
  }
  // D layout: VGPR r -> M = r + 8*(lane>=16), N = lane&15
  float* crow = C + (m0 + hi8) * (long)O + n0 + rc;
#pragma unroll
  for (int t = 0; t < 4; ++t)
#pragma unroll
    for (int r = 0; r < 8; ++r)
      crow[(long)r * O + t * 16] = acc[t][r];
}

// ---------------------------------------------------------------- aggregation
// agg = dinv[i]^2 * hw[i]   (self-loop contribution seeds the accumulator)
__global__ void self_init(const float* __restrict__ hw, const float* __restrict__ dinv,
                          float* agg, int O, long total) {
  long i = (long)blockIdx.x * blockDim.x + threadIdx.x;
  if (i >= total) return;
  float di = dinv[i / O];
  agg[i] = di * di * hw[i];
}

// one wave per edge: agg[dst] += norm[e] * hw[src]
__global__ void scatter_edges(const long long* __restrict__ ei,
                              const float* __restrict__ nrm,
                              const float* __restrict__ hw,
                              float* agg, int E, int O) {
  long t = (long)blockIdx.x * blockDim.x + threadIdx.x;
  int e = (int)(t >> 5);
  int lane = (int)(t & 31);
  if (e >= E) return;
  int src = (int)ei[e];                // edge_index[0] = source (col)
  int dst = (int)ei[(long)E + e];      // edge_index[1] = target (row)
  float w = nrm[e];
  const float* s = hw + (long)src * O;
  float* d = agg + (long)dst * O;
  for (int f = lane; f < O; f += 32) atomicAdd(&d[f], w * s[f]);
}

// out = relu( (a + bias - m) * rsqrt(v+eps) * g + be )
__global__ void bn_relu(const float* __restrict__ a, const float* __restrict__ bias,
                        const float* __restrict__ g, const float* __restrict__ be,
                        const float* __restrict__ m, const float* __restrict__ v,
                        float* out, int O, long total) {
  long i = (long)blockIdx.x * blockDim.x + threadIdx.x;
  if (i >= total) return;
  int f = (int)(i % O);
  float xv = a[i] + bias[f];
  out[i] = fmaxf((xv - m[f]) * rsqrtf(v[f] + 1e-5f) * g[f] + be[f], 0.0f);
}

// ---------------------------------------------------------------- pooling
__global__ void pool_accum(const float* __restrict__ h, const long long* __restrict__ batch,
                           float* pooled, float* counts, int N, int O) {
  long t = (long)blockIdx.x * blockDim.x + threadIdx.x;
  int node = (int)(t >> 5);
  int lane = (int)(t & 31);
  if (node >= N) return;
  int b = (int)batch[node];
  if (lane == 0) atomicAdd(&counts[b], 1.0f);
  const float* s = h + (long)node * O;
  float* d = pooled + (long)b * O;
  for (int f = lane; f < O; f += 32) atomicAdd(&d[f], s[f]);
}

__global__ void pool_div(float* pooled, const float* __restrict__ counts, int O, long total) {
  long i = (long)blockIdx.x * blockDim.x + threadIdx.x;
  if (i >= total) return;
  pooled[i] /= fmaxf(counts[i / O], 1.0f);
}

// ---------------------------------------------------------------- final FC (C=10)
__global__ void final_fc(const float* __restrict__ z1, const float* __restrict__ Wl2,
                         const float* __restrict__ bl2, float* out, int K, int C) {
  int b = blockIdx.x;
  int c = threadIdx.x;
  if (c >= C) return;
  const float* zr = z1 + (long)b * K;
  float acc = bl2[c];
  for (int k = 0; k < K; ++k) acc = fmaf(zr[k], Wl2[(long)k * C + c], acc);
  out[(long)b * C + c] = acc;
}

// ---------------------------------------------------------------- launch
extern "C" void kernel_launch(void* const* d_in, const int* in_sizes, int n_in,
                              void* d_out, int out_size, void* d_ws, size_t ws_size,
                              hipStream_t stream) {
  const float*     x     = (const float*)d_in[0];
  const long long* ei    = (const long long*)d_in[1];   // int64 [2,E] flattened
  const long long* batch = (const long long*)d_in[2];   // int64 [N]
  // d_in[3] = batch_size scalar (known: 256)
  const float *W1 = (const float*)d_in[4],  *b1 = (const float*)d_in[5];
  const float *g1 = (const float*)d_in[6],  *be1= (const float*)d_in[7];
  const float *m1 = (const float*)d_in[8],  *v1 = (const float*)d_in[9];
  const float *W2 = (const float*)d_in[10], *b2 = (const float*)d_in[11];
  const float *g2 = (const float*)d_in[12], *be2= (const float*)d_in[13];
  const float *m2 = (const float*)d_in[14], *v2 = (const float*)d_in[15];
  const float *W3 = (const float*)d_in[16], *b3 = (const float*)d_in[17];
  const float *g3 = (const float*)d_in[18], *be3= (const float*)d_in[19];
  const float *m3 = (const float*)d_in[20], *v3 = (const float*)d_in[21];
  const float *Wl1= (const float*)d_in[22], *bl1= (const float*)d_in[23];
  const float *g4 = (const float*)d_in[24], *be4= (const float*)d_in[25];
  const float *m4 = (const float*)d_in[26], *v4 = (const float*)d_in[27];
  const float *Wl2= (const float*)d_in[28], *bl2= (const float*)d_in[29];

  const int N = in_sizes[0] / 128;   // 50000 (multiple of 16)
  const int E = in_sizes[1] / 2;     // 800000
  const int B = 256;

  // workspace layout
  float* deg    = (float*)d_ws;                 // N
  float* dinv   = deg    + N;                   // N
  float* nrm    = dinv   + N;                   // E
  float* hw     = nrm    + E;                   // N*256 (GEMM out / z1raw)
  float* agg    = hw     + (long)N * 256;       // N*256
  float* hbuf   = agg    + (long)N * 256;       // N*256 (layer activations, f32)
  float* pooled = hbuf   + (long)N * 256;       // B*256
  float* counts = pooled + (long)B * 256;       // B
  float* z1     = counts + B;                   // B*512
  _Float16* ah  = (_Float16*)(z1 + (long)B * 512);  // N*256 halves (packed A)
  _Float16* wt1 = ah   + (long)N * 256;         // 64*128
  _Float16* wt2 = wt1  + 64 * 128;              // 128*64
  _Float16* wt3 = wt2  + 128 * 64;              // 256*128
  _Float16* wtl1= wt3  + 256 * 128;             // 512*256
  _Float16* ph  = wtl1 + 512 * 256;             // B*256 (packed pooled)

  auto cdiv = [](long a, long b) { return (unsigned)((a + b - 1) / b); };

  // ---- pre-pack all weights (transposed f16)
  pack_wt<<<cdiv(128 * 64, 256), 256, 0, stream>>>(W1, wt1, 128, 64);
  pack_wt<<<cdiv(64 * 128, 256), 256, 0, stream>>>(W2, wt2, 64, 128);
  pack_wt<<<cdiv(128 * 256, 256), 256, 0, stream>>>(W3, wt3, 128, 256);
  pack_wt<<<cdiv(256 * 512, 256), 256, 0, stream>>>(Wl1, wtl1, 256, 512);

  // ---- degrees and GCN edge norms
  fill_f32 <<<cdiv(N, 256), 256, 0, stream>>>(deg, 1.0f, N);
  deg_accum<<<cdiv(E, 256), 256, 0, stream>>>(ei, deg, E);
  rsqrt_k  <<<cdiv(N, 256), 256, 0, stream>>>(deg, dinv, N);
  edge_norm<<<cdiv(E, 256), 256, 0, stream>>>(ei, dinv, nrm, E);

  // ---- one GCN layer: pack -> wmma gemm -> self seed -> scatter -> bias+BN+ReLU
  auto layer = [&](const float* hin, int K, int O, const _Float16* Wt, const float* bias,
                   const float* g, const float* be, const float* m, const float* v,
                   float* hout) {
    pack_f16<<<cdiv((long)N * K, 256), 256, 0, stream>>>(hin, ah, (long)N * K);
    dim3 gg(N / 16, O / 64);
    gemm_wmma_packed<<<gg, 32, 0, stream>>>(ah, Wt, hw, K, O);
    long total = (long)N * O;
    self_init    <<<cdiv(total, 256),        256, 0, stream>>>(hw, dinv, agg, O, total);
    scatter_edges<<<cdiv((long)E * 32, 256), 256, 0, stream>>>(ei, nrm, hw, agg, E, O);
    bn_relu      <<<cdiv(total, 256),        256, 0, stream>>>(agg, bias, g, be, m, v,
                                                               hout, O, total);
  };
  layer(x,    128,  64, wt1, b1, g1, be1, m1, v1, hbuf);
  layer(hbuf,  64, 128, wt2, b2, g2, be2, m2, v2, hbuf);
  layer(hbuf, 128, 256, wt3, b3, g3, be3, m3, v3, hbuf);

  // ---- global mean pool over graphs
  fill_f32  <<<cdiv((long)B * 256, 256), 256, 0, stream>>>(pooled, 0.0f, (long)B * 256);
  fill_f32  <<<1, 256, 0, stream>>>(counts, 0.0f, B);
  pool_accum<<<cdiv((long)N * 32, 256), 256, 0, stream>>>(hbuf, batch, pooled, counts, N, 256);
  pool_div  <<<cdiv((long)B * 256, 256), 256, 0, stream>>>(pooled, counts, 256, (long)B * 256);

  // ---- head: z1 = relu(bn(pooled @ Wl1 + bl1)); out = z1 @ Wl2 + bl2
  pack_f16<<<cdiv((long)B * 256, 256), 256, 0, stream>>>(pooled, ph, (long)B * 256);
  dim3 gl(B / 16, 512 / 64);
  gemm_wmma_packed<<<gl, 32, 0, stream>>>(ph, wtl1, hw, 256, 512);
  bn_relu<<<cdiv((long)B * 512, 256), 256, 0, stream>>>(hw, bl1, g4, be4, m4, v4,
                                                        z1, 512, (long)B * 512);
  final_fc<<<B, 32, 0, stream>>>(z1, Wl2, bl2, (float*)d_out, 512, 10);
}
